// Attention_73083163509159
// MI455X (gfx1250) — matrix operand order repaired and checked by reading, not compiled, not run on hardware
//
#include <hip/hip_runtime.h>

#define H      8
#define DH     64
#define BATCH  2
#define SEQ    2048
#define DMODEL 256
#define INNER  512
#define MTOT   (BATCH*SEQ)   // 4096

typedef __attribute__((ext_vector_type(16))) __bf16 v16bf;
typedef __attribute__((ext_vector_type(8)))  float  v8f;
typedef __attribute__((ext_vector_type(4)))  int    vi4;

typedef __attribute__((address_space(1))) vi4 gvi4;   // global int4
typedef __attribute__((address_space(3))) vi4 lvi4;   // LDS int4

union Frag16 { v16bf v; uint4 u[2]; };

__device__ __forceinline__ unsigned short f2bf(float f) {
  unsigned int x = __float_as_uint(f);
  unsigned int r = x + 0x7FFFu + ((x >> 16) & 1u);   // round-to-nearest-even
  return (unsigned short)(r >> 16);
}

__device__ __forceinline__ float red_max16(float v) {
  v = fmaxf(v, __shfl_xor(v, 1, 32));
  v = fmaxf(v, __shfl_xor(v, 2, 32));
  v = fmaxf(v, __shfl_xor(v, 4, 32));
  v = fmaxf(v, __shfl_xor(v, 8, 32));
  return v;
}
__device__ __forceinline__ float red_sum16(float v) {
  v += __shfl_xor(v, 1, 32);
  v += __shfl_xor(v, 2, 32);
  v += __shfl_xor(v, 4, 32);
  v += __shfl_xor(v, 8, 32);
  return v;
}

// ---- CDNA5 async global->LDS copy (16B per lane), ASYNCcnt-tracked ----
__device__ __forceinline__ void async_ld16(const unsigned short* gsrc,
                                           unsigned short* ldst) {
#if __has_builtin(__builtin_amdgcn_global_load_async_to_lds_b128)
  __builtin_amdgcn_global_load_async_to_lds_b128(
      (gvi4*)gsrc, (lvi4*)ldst, 0, 0);
#else
  unsigned loff = (unsigned)(size_t)(__attribute__((address_space(3))) unsigned short*)ldst;
  asm volatile("global_load_async_to_lds_b128 %0, %1, off"
               :: "v"(loff), "v"(gsrc) : "memory");
#endif
}

__device__ __forceinline__ void wait_async0() {
#if __has_builtin(__builtin_amdgcn_s_wait_asynccnt)
  __builtin_amdgcn_s_wait_asynccnt(0);
#else
  asm volatile("s_wait_asynccnt 0x0" ::: "memory");
#endif
}

// ---------------- prep kernels ----------------
__global__ void cvt_bf16_kernel(const float* __restrict__ in,
                                unsigned short* __restrict__ out, int n) {
  for (int i = blockIdx.x * blockDim.x + threadIdx.x; i < n; i += gridDim.x * blockDim.x)
    out[i] = f2bf(in[i]);
}

// outT[c][r] = bf16(in[r][c])
__global__ void transpose_bf16_kernel(const float* __restrict__ in,
                                      unsigned short* __restrict__ outT,
                                      int rows, int cols) {
  int total = rows * cols;
  for (int i = blockIdx.x * blockDim.x + threadIdx.x; i < total; i += gridDim.x * blockDim.x) {
    int r = i / cols, c = i - r * cols;
    outT[c * rows + r] = f2bf(in[i]);
  }
}

// ---------------- projection GEMM: X(4096x256) @ W(256x512) ----------------
// modes: 0 -> q  (scaled, bf16 [b][h][n][dh])
//        1 -> k  (bf16 [b][h][n][dh])
//        2 -> vT (bf16 [b][h][dh][n])
//        3 -> gates = sigmoid(v + bg) (f32 [m][512])
__global__ void __launch_bounds__(256)
proj_gemm_kernel(const unsigned short* __restrict__ xbf,
                 const unsigned short* __restrict__ wT,   // [512][256] bf16 (W transposed)
                 const float* __restrict__ bvec,
                 unsigned short* __restrict__ obf,
                 float* __restrict__ of32,
                 int mode, float scale)
{
  const int lane = threadIdx.x & 31;
  const int wave = threadIdx.x >> 5;
  const int half = lane >> 4;
  const int l16  = lane & 15;
  const int m0 = blockIdx.x * 16;
  const int n0 = blockIdx.y * 128 + wave * 16;

  v8f acc = {};
  const unsigned short* arow = xbf + (m0 + l16) * DMODEL;
  const unsigned short* brow = wT + (n0 + l16) * DMODEL;
#pragma unroll
  for (int k0 = 0; k0 < DMODEL; k0 += 32) {
    Frag16 a, b;
    a.u[0] = *(const uint4*)(arow + k0 + 8 * half);
    a.u[1] = *(const uint4*)(arow + k0 + 16 + 8 * half);
    b.u[0] = *(const uint4*)(brow + k0 + 8 * half);
    b.u[1] = *(const uint4*)(brow + k0 + 16 + 8 * half);
    acc = __builtin_amdgcn_wmma_f32_16x16x32_bf16(false, a.v, false, b.v,
                                                  (short)0, acc, false, false);
  }
#pragma unroll
  for (int r = 0; r < 8; ++r) {
    int m = m0 + r + 8 * half;          // global row = b*SEQ + seq
    int c = n0 + l16;                   // 0..511
    float v = acc[r];
    if (mode == 3) {
      v += bvec[c];
      of32[m * INNER + c] = 1.0f / (1.0f + __expf(-v));
    } else {
      int b  = m >> 11;
      int s  = m & (SEQ - 1);
      int h  = c >> 6;
      int dh = c & 63;
      if (mode == 0) v *= scale;
      int idx;
      if (mode == 2) idx = ((b * H + h) * DH + dh) * SEQ + s;   // V transposed
      else           idx = ((b * H + h) * SEQ + s) * DH + dh;
      obf[idx] = f2bf(v);
    }
  }
}

// ---------------- fused flash attention ----------------
// grid: BATCH*H*(SEQ/128) blocks of 256 threads; wave w owns queries [q0, q0+16).
// All 8 waves share (b,h): K/V tiles are staged into LDS once per block with
// CDNA5 async global->LDS copies, double-buffered against the WMMA pipeline.
__global__ void __launch_bounds__(256)
flash_attn_kernel(const unsigned short* __restrict__ qb,
                  const unsigned short* __restrict__ kb,
                  const unsigned short* __restrict__ vtb,
                  const float* __restrict__ bias,
                  const unsigned char* __restrict__ mask,
                  const float* __restrict__ gates,
                  unsigned short* __restrict__ aout)
{
  __shared__ alignas(16) unsigned short kTile[2][32 * DH];   // [key][dh]  4KB x2
  __shared__ alignas(16) unsigned short vTile[2][DH * 32];   // [dh][key]  4KB x2
  __shared__ alignas(16) unsigned short pTile[8][16 * 32];   // per-wave P  8KB

  const int tid  = threadIdx.x;
  const int lane = tid & 31;
  const int wave = tid >> 5;
  const int half = lane >> 4;
  const int l16  = lane & 15;

  const int bh   = blockIdx.x >> 4;          // b*H + h
  const int qgrp = blockIdx.x & 15;
  const int b    = bh >> 3;
  const int h    = bh & 7;
  const int q0   = (qgrp * 8 + wave) * 16;   // seq index of first query row

  const unsigned short* qp  = qb  + bh * SEQ * DH;
  const unsigned short* kp  = kb  + bh * SEQ * DH;
  const unsigned short* vtp = vtb + bh * DH * SEQ;
  const float* bp = bias + (size_t)bh * SEQ * SEQ;
  const unsigned char* mkp = mask + b * SEQ;

  // per-thread staging assignment (16B each; 256 threads cover 4KB per tile)
  const int vr = tid >> 2;        // dh row 0..63
  const int vs = tid & 3;         // 8-element segment 0..3

  // Q A-fragments for DH contraction steps k0 = 0 and 32
  Frag16 aq0, aq1;
  {
    const unsigned short* qrow = qp + (q0 + l16) * DH;
    aq0.u[0] = *(const uint4*)(qrow + 0  + 8 * half);
    aq0.u[1] = *(const uint4*)(qrow + 16 + 8 * half);
    aq1.u[0] = *(const uint4*)(qrow + 32 + 8 * half);
    aq1.u[1] = *(const uint4*)(qrow + 48 + 8 * half);
  }

  unsigned char qm[8];
  float m_run[8], l_run[8];
#pragma unroll
  for (int r = 0; r < 8; ++r) {
    qm[r]    = mkp[q0 + r + 8 * half];
    m_run[r] = -3.402823466e38f;
    l_run[r] = 0.0f;
  }
  v8f zero = {};
  v8f accO[4] = { zero, zero, zero, zero };   // 4 dh-tiles of 16x16

  // prologue: stage tile 0
  async_ld16(kp + tid * 8,                 &kTile[0][tid * 8]);
  async_ld16(vtp + vr * SEQ + vs * 8,      &vTile[0][vr * 32 + vs * 8]);
  wait_async0();
  __syncthreads();

  int cur = 0;
  for (int kbk = 0; kbk < SEQ; kbk += 32) {
    // stage next K/V tile into the other buffer while we compute this one
    if (kbk + 32 < SEQ) {
      const int nxt = kbk + 32;
      async_ld16(kp + nxt * DH + tid * 8,        &kTile[cur ^ 1][tid * 8]);
      async_ld16(vtp + vr * SEQ + nxt + vs * 8,  &vTile[cur ^ 1][vr * 32 + vs * 8]);
    }

    // ---- S = Q @ K^T for 32 keys (two 16-key sub-tiles), K from LDS ----
    v8f s[2];
#pragma unroll
    for (int sub = 0; sub < 2; ++sub) {
      const unsigned short* krow = &kTile[cur][(sub * 16 + l16) * DH];
      Frag16 b0, b1;
      b0.u[0] = *(const uint4*)(krow + 0  + 8 * half);
      b0.u[1] = *(const uint4*)(krow + 16 + 8 * half);
      b1.u[0] = *(const uint4*)(krow + 32 + 8 * half);
      b1.u[1] = *(const uint4*)(krow + 48 + 8 * half);
      v8f acc = zero;
      acc = __builtin_amdgcn_wmma_f32_16x16x32_bf16(false, aq0.v, false, b0.v,
                                                    (short)0, acc, false, false);
      acc = __builtin_amdgcn_wmma_f32_16x16x32_bf16(false, aq1.v, false, b1.v,
                                                    (short)0, acc, false, false);
      s[sub] = acc;
    }

    const unsigned char km0 = mkp[kbk + l16];
    const unsigned char km1 = mkp[kbk + 16 + l16];

    // ---- bias (streamed non-temporal), mask, online softmax ----
    float p0[8], p1[8], alpha[8];
#pragma unroll
    for (int r = 0; r < 8; ++r) {
      const int mrow = q0 + r + 8 * half;
      const float* brw = bp + (size_t)mrow * SEQ + kbk;
      float s0 = s[0][r] + __builtin_nontemporal_load(brw + l16);
      float s1 = s[1][r] + __builtin_nontemporal_load(brw + 16 + l16);
      s0 = (qm[r] && km0) ? s0 : -3.402823466e38f;
      s1 = (qm[r] && km1) ? s1 : -3.402823466e38f;
      float rmax = red_max16(fmaxf(s0, s1));
      float mnew = fmaxf(m_run[r], rmax);
      float a    = __expf(m_run[r] - mnew);
      m_run[r]   = mnew;
      alpha[r]   = a;
      p0[r] = __expf(s0 - mnew);
      p1[r] = __expf(s1 - mnew);
      float rsum = red_sum16(p0[r] + p1[r]);
      l_run[r]   = l_run[r] * a + rsum;
    }

    // rescale running output
#pragma unroll
    for (int t = 0; t < 4; ++t)
#pragma unroll
      for (int r = 0; r < 8; ++r)
        accO[t][r] *= alpha[r];

    // ---- transpose P (C-layout -> A-layout) via per-wave LDS tile ----
    unsigned short* pw = &pTile[wave][0];
#pragma unroll
    for (int r = 0; r < 8; ++r) {
      const int row = r + 8 * half;
      pw[row * 32 + l16]      = f2bf(p0[r]);
      pw[row * 32 + 16 + l16] = f2bf(p1[r]);
    }
    Frag16 ap;
    {
      const unsigned short* prow = pw + l16 * 32;
      ap.u[0] = *(const uint4*)(prow + 16 * half);
      ap.u[1] = *(const uint4*)(prow + 16 * half + 8);
    }

    // ---- O += P @ V  (V^T tile from LDS: contiguous B-fragments) ----
#pragma unroll
    for (int t = 0; t < 4; ++t) {
      const unsigned short* vrow = &vTile[cur][(t * 16 + l16) * 32 + 16 * half];
      Frag16 bv;
      bv.u[0] = *(const uint4*)(vrow);
      bv.u[1] = *(const uint4*)(vrow + 8);
      accO[t] = __builtin_amdgcn_wmma_f32_16x16x32_bf16(false, ap.v, false, bv.v,
                                                        (short)0, accO[t], false, false);
    }

    // publish next tile: own async copies done, then block-wide barrier
    wait_async0();
    __syncthreads();
    cur ^= 1;
  }

  // ---- epilogue: normalize, apply sigmoid gate, store bf16 [m][h*64+dh] ----
#pragma unroll
  for (int t = 0; t < 4; ++t)
#pragma unroll
    for (int r = 0; r < 8; ++r) {
      const int mrow = q0 + r + 8 * half;
      const int gm = b * SEQ + mrow;
      const int c  = h * DH + t * 16 + l16;
      float v = accO[t][r] / l_run[r];
      v *= gates[gm * INNER + c];
      aout[gm * INNER + c] = f2bf(v);
    }
}

// ---------------- output GEMM: (O*g)(4096x512) @ Wout(512x256) + bout ----------------
__global__ void __launch_bounds__(256)
out_gemm_kernel(const unsigned short* __restrict__ aout,
                const unsigned short* __restrict__ woutT,  // [256][512] bf16
                const float* __restrict__ bout,
                float* __restrict__ out)
{
  const int lane = threadIdx.x & 31;
  const int wave = threadIdx.x >> 5;
  const int half = lane >> 4;
  const int l16  = lane & 15;
  const int m0 = blockIdx.x * 16;
  const int n0 = blockIdx.y * 128 + wave * 16;

  v8f acc = {};
  const unsigned short* arow = aout  + (m0 + l16) * INNER;
  const unsigned short* brow = woutT + (n0 + l16) * INNER;
#pragma unroll
  for (int k0 = 0; k0 < INNER; k0 += 32) {
    Frag16 a, b;
    a.u[0] = *(const uint4*)(arow + k0 + 8 * half);
    a.u[1] = *(const uint4*)(arow + k0 + 16 + 8 * half);
    b.u[0] = *(const uint4*)(brow + k0 + 8 * half);
    b.u[1] = *(const uint4*)(brow + k0 + 16 + 8 * half);
    acc = __builtin_amdgcn_wmma_f32_16x16x32_bf16(false, a.v, false, b.v,
                                                  (short)0, acc, false, false);
  }
#pragma unroll
  for (int r = 0; r < 8; ++r) {
    const int m = m0 + r + 8 * half;
    const int n = n0 + l16;
    out[m * DMODEL + n] = acc[r] + bout[n];
  }
}

extern "C" void kernel_launch(void* const* d_in, const int* in_sizes, int n_in,
                              void* d_out, int out_size, void* d_ws, size_t ws_size,
                              hipStream_t stream)
{
  (void)in_sizes; (void)n_in; (void)out_size; (void)ws_size;
  const float*         x         = (const float*)d_in[0];
  const unsigned char* mask      = (const unsigned char*)d_in[1];
  const float*         attn_bias = (const float*)d_in[2];
  const float*         Wq        = (const float*)d_in[3];
  const float*         Wkv       = (const float*)d_in[4];
  const float*         Wout      = (const float*)d_in[5];
  const float*         bout      = (const float*)d_in[6];
  const float*         Wg        = (const float*)d_in[7];
  const float*         bg        = (const float*)d_in[8];
  float* out = (float*)d_out;

  char* ws = (char*)d_ws;
  size_t off = 0;
  auto alloc = [&](size_t bytes) -> void* {
    void* p = ws + off;
    off += (bytes + 255) & ~(size_t)255;
    return p;
  };
  unsigned short* xbf   = (unsigned short*)alloc((size_t)MTOT * DMODEL * 2);
  unsigned short* WqT   = (unsigned short*)alloc((size_t)INNER * DMODEL * 2);
  unsigned short* WkvT  = (unsigned short*)alloc((size_t)2 * INNER * DMODEL * 2);
  unsigned short* WgT   = (unsigned short*)alloc((size_t)INNER * DMODEL * 2);
  unsigned short* WoutT = (unsigned short*)alloc((size_t)DMODEL * INNER * 2);
  unsigned short* qb    = (unsigned short*)alloc((size_t)BATCH * H * SEQ * DH * 2);
  unsigned short* kbuf  = (unsigned short*)alloc((size_t)BATCH * H * SEQ * DH * 2);
  unsigned short* vtb   = (unsigned short*)alloc((size_t)BATCH * H * SEQ * DH * 2);
  float*          gates = (float*)alloc((size_t)MTOT * INNER * 4);
  unsigned short* abuf  = (unsigned short*)alloc((size_t)MTOT * INNER * 2);

  // prep: bf16 x, transposed bf16 weights
  cvt_bf16_kernel<<<(MTOT * DMODEL + 255) / 256, 256, 0, stream>>>(x, xbf, MTOT * DMODEL);
  transpose_bf16_kernel<<<(DMODEL * INNER     + 255) / 256, 256, 0, stream>>>(Wq,   WqT,   DMODEL, INNER);
  transpose_bf16_kernel<<<(DMODEL * 2 * INNER + 255) / 256, 256, 0, stream>>>(Wkv,  WkvT,  DMODEL, 2 * INNER);
  transpose_bf16_kernel<<<(DMODEL * INNER     + 255) / 256, 256, 0, stream>>>(Wg,   WgT,   DMODEL, INNER);
  transpose_bf16_kernel<<<(INNER * DMODEL     + 255) / 256, 256, 0, stream>>>(Wout, WoutT, INNER, DMODEL);

  // projections: q (scaled), k, vT, gates
  dim3 pg(MTOT / 16, INNER / 128);
  proj_gemm_kernel<<<pg, 256, 0, stream>>>(xbf, WqT,                   nullptr, qb,      nullptr, 0, 0.125f);
  proj_gemm_kernel<<<pg, 256, 0, stream>>>(xbf, WkvT,                  nullptr, kbuf,    nullptr, 1, 1.0f);
  proj_gemm_kernel<<<pg, 256, 0, stream>>>(xbf, WkvT + INNER * DMODEL, nullptr, vtb,     nullptr, 2, 1.0f);
  proj_gemm_kernel<<<pg, 256, 0, stream>>>(xbf, WgT,                   bg,      nullptr, gates,   3, 1.0f);

  // fused flash attention (streams the 268 MB bias once, non-temporal;
  // K/V tiles double-buffered in LDS via async global->LDS copies)
  flash_attn_kernel<<<BATCH * H * (SEQ / 128), 256, 0, stream>>>(
      qb, kbuf, vtb, attn_bias, mask, gates, abuf);

  // output projection + bias
  dim3 og(MTOT / 16, DMODEL / 128);
  out_gemm_kernel<<<og, 256, 0, stream>>>(abuf, WoutT, bout, out);
}
